// SS2D_42417097015430
// MI455X (gfx1250) — matrix operand order repaired and checked by reading, not compiled, not run on hardware
//
#include <hip/hip_runtime.h>
#include <hip/hip_bf16.h>
#include <math.h>

typedef __attribute__((ext_vector_type(16))) __bf16 v16bf;
typedef __attribute__((ext_vector_type(8)))  float  v8f;

// ---------------- problem constants ----------------
constexpr int cB  = 32;
constexpr int cH  = 15;
constexpr int cW  = 15;
constexpr int cC  = 256;
constexpr int cDI = 512;           // d_inner
constexpr int cN  = 16;            // d_state
constexpr int cL  = 225;           // H*W
constexpr int cBL = cB * cL;       // 7200
constexpr int cBD = cB * cDI;      // 16384

__device__ __forceinline__ float sigf(float x) { return 1.0f / (1.0f + __expf(-x)); }
__device__ __forceinline__ float siluf(float x) { return x * sigf(x); }

// =====================================================================
// Generic WMMA GEMM:  C[M,N] = A[M,K] (row-major, lda) * W[N,K]^T (row-major, ldw)
// fp32 in/out, bf16 multiply via v_wmma_f32_16x16x32_bf16.
// KT is compile-time K; AV/BV enable float4 (b128) fragment loads when the
// corresponding stride is a multiple of 4 floats.  All fragment fills are
// branchless: the ragged K tail uses clamped indices + value selects, and
// out-of-range output columns read a clamped W row (their lanes only feed
// columns that are never stored).
// =====================================================================
template<int KT, bool AV, bool BV>
__global__ void wmma_gemm_kernel(const float* __restrict__ A, int lda,
                                 const float* __restrict__ Wt, int ldw,
                                 float* __restrict__ C, int ldc,
                                 int M, int N)
{
    const int lane = threadIdx.x & 31;
    const int wave = threadIdx.x >> 5;
    const int m0 = (blockIdx.y * 4 + wave) * 16;
    const int n0 = blockIdx.x * 16;
    if (m0 >= M) return;                    // wave-uniform: EXEC stays all-ones

    const int ml  = lane & 15;
    const int kgA = (lane & 16) ? 8 : 0;    // A: 16-bit 16x32 layout
    const int kgB = (lane & 16) ? 8 : 0;    // B: 16-bit 16x32 layout (same K order as A)

    const float* __restrict__ Ar = A + (size_t)(m0 + ml) * lda;   // M tiles 16-aligned
    int wr = n0 + ml; if (wr >= N) wr = N - 1;                    // clamp, never stored
    const float* __restrict__ Wr = Wt + (size_t)wr * ldw;

    v8f acc = {};
    constexpr int KFULL = (KT / 32) * 32;

    for (int k0 = 0; k0 < KFULL; k0 += 32) {
        v16bf af, bf;
        if constexpr (AV) {
            const float4* p0 = reinterpret_cast<const float4*>(Ar + k0 + kgA);
            const float4* p1 = reinterpret_cast<const float4*>(Ar + k0 + kgA + 16);
            float4 a0 = p0[0], a1 = p0[1], a2 = p1[0], a3 = p1[1];
            af[0]=(__bf16)a0.x; af[1]=(__bf16)a0.y; af[2]=(__bf16)a0.z; af[3]=(__bf16)a0.w;
            af[4]=(__bf16)a1.x; af[5]=(__bf16)a1.y; af[6]=(__bf16)a1.z; af[7]=(__bf16)a1.w;
            af[8]=(__bf16)a2.x; af[9]=(__bf16)a2.y; af[10]=(__bf16)a2.z; af[11]=(__bf16)a2.w;
            af[12]=(__bf16)a3.x; af[13]=(__bf16)a3.y; af[14]=(__bf16)a3.z; af[15]=(__bf16)a3.w;
        } else {
            #pragma unroll
            for (int e = 0; e < 16; ++e)
                af[e] = (__bf16)Ar[k0 + kgA + ((e & 8) << 1) + (e & 7)];
        }
        if constexpr (BV) {
            const float4* p = reinterpret_cast<const float4*>(Wr + k0 + kgB);
            float4 b0 = p[0], b1 = p[1], b2 = p[4], b3 = p[5];
            bf[0]=(__bf16)b0.x; bf[1]=(__bf16)b0.y; bf[2]=(__bf16)b0.z; bf[3]=(__bf16)b0.w;
            bf[4]=(__bf16)b1.x; bf[5]=(__bf16)b1.y; bf[6]=(__bf16)b1.z; bf[7]=(__bf16)b1.w;
            bf[8]=(__bf16)b2.x; bf[9]=(__bf16)b2.y; bf[10]=(__bf16)b2.z; bf[11]=(__bf16)b2.w;
            bf[12]=(__bf16)b3.x; bf[13]=(__bf16)b3.y; bf[14]=(__bf16)b3.z; bf[15]=(__bf16)b3.w;
        } else {
            #pragma unroll
            for (int e = 0; e < 16; ++e)
                bf[e] = (__bf16)Wr[k0 + kgB + ((e & 8) << 1) + (e & 7)];
        }
        acc = __builtin_amdgcn_wmma_f32_16x16x32_bf16(
            false, af, false, bf, (short)0, acc, false, false);
    }

    if constexpr ((KT % 32) != 0) {
        // ragged tail: branchless clamped loads; zero only A's out-of-range
        // K slots (each k-product pairs A and B at the same K position).
        v16bf af, bf;
        #pragma unroll
        for (int e = 0; e < 16; ++e) {
            int ka = KFULL + kgA + ((e & 8) << 1) + (e & 7);
            int kb = KFULL + kgB + ((e & 8) << 1) + (e & 7);
            float av = Ar[ka < KT ? ka : 0];
            af[e] = (ka < KT) ? (__bf16)av : (__bf16)0.0f;
            bf[e] = (__bf16)Wr[kb < KT ? kb : 0];
        }
        acc = __builtin_amdgcn_wmma_f32_16x16x32_bf16(
            false, af, false, bf, (short)0, acc, false, false);
    }

    const int mbase = m0 + ((lane & 16) ? 8 : 0);
    const int col = n0 + ml;
    if (col < N) {
        #pragma unroll
        for (int r = 0; r < 8; ++r)
            C[(size_t)(mbase + r) * ldc + col] = acc[r];
    }
}

// =====================================================================
// Depthwise 3x3 conv (pad 1) + bias + SiLU.  Input: x part (cols [0,DI)) of
// xz laid out (B*L, 2*DI).  Output: (B*L, DI).
// =====================================================================
__global__ void dwconv_silu_kernel(const float* __restrict__ xz,
                                   const float* __restrict__ cw,   // (DI,9)
                                   const float* __restrict__ cb,   // (DI)
                                   float* __restrict__ out)
{
    int i = blockIdx.x * blockDim.x + threadIdx.x;
    if (i >= cBL * cDI) return;
    int d = i % cDI;
    int t = (i / cDI) % cL;
    int b = i / (cDI * cL);
    int h = t / cW, w = t % cW;
    float acc = cb[d];
    #pragma unroll
    for (int dh = 0; dh < 3; ++dh) {
        int hh = h + dh - 1; if (hh < 0 || hh >= cH) continue;
        #pragma unroll
        for (int dw = 0; dw < 3; ++dw) {
            int ww = w + dw - 1; if (ww < 0 || ww >= cW) continue;
            acc += xz[((size_t)(b * cL + hh * cW + ww)) * (2 * cDI) + d]
                 * cw[d * 9 + dh * 3 + dw];
        }
    }
    out[i] = siluf(acc);
}

// ---------- spatial adaptive attention ----------
__global__ void spa_meanmax_kernel(const float* __restrict__ xc, float* __restrict__ mm)
{
    int row = blockIdx.x;                 // (b,t)
    int b = row / cL, t = row % cL;
    __shared__ float sm[256], sx[256];
    float m = 0.f, x = -1e30f;
    for (int d = threadIdx.x; d < cDI; d += 256) {
        float v = xc[(size_t)row * cDI + d];
        m += v; x = fmaxf(x, v);
    }
    sm[threadIdx.x] = m; sx[threadIdx.x] = x;
    __syncthreads();
    for (int s = 128; s > 0; s >>= 1) {
        if (threadIdx.x < s) {
            sm[threadIdx.x] += sm[threadIdx.x + s];
            sx[threadIdx.x] = fmaxf(sx[threadIdx.x], sx[threadIdx.x + s]);
        }
        __syncthreads();
    }
    if (threadIdx.x == 0) {
        mm[(size_t)b * 2 * cL + t]      = sm[0] / cDI;
        mm[(size_t)b * 2 * cL + cL + t] = sx[0];
    }
}

__global__ void spa_apply_kernel(const float* __restrict__ mm,
                                 const float* __restrict__ wc,   // (2,3,3)
                                 const float* __restrict__ aw,   // (L)
                                 float* __restrict__ xc)
{
    int row = blockIdx.x;
    int b = row / cL, t = row % cL;
    int h = t / cW, w = t % cW;
    __shared__ float a2s;
    if (threadIdx.x == 0) {
        float s = 0.f;
        for (int c = 0; c < 2; ++c)
            for (int dh = 0; dh < 3; ++dh) {
                int hh = h + dh - 1; if (hh < 0 || hh >= cH) continue;
                for (int dw = 0; dw < 3; ++dw) {
                    int ww = w + dw - 1; if (ww < 0 || ww >= cW) continue;
                    s += mm[(size_t)b * 2 * cL + c * cL + hh * cW + ww]
                       * wc[c * 9 + dh * 3 + dw];
                }
            }
        float att = sigf(s);
        a2s = sigf(att * (1.0f + aw[t]));
    }
    __syncthreads();
    float a2 = a2s;
    for (int d = threadIdx.x; d < cDI; d += blockDim.x)
        xc[(size_t)row * cDI + d] *= a2;
}

// ---------- softplus(X + bias[col]) in place ----------
__global__ void softplus_bias_kernel(float* __restrict__ X, const float* __restrict__ bias,
                                     int rows, int cols)
{
    int i = blockIdx.x * blockDim.x + threadIdx.x;
    if (i >= rows * cols) return;
    float v = X[i] + bias[i % cols];
    X[i] = (v > 20.0f) ? v : log1pf(__expf(v));
}

// ---------- spatial selective scan: one thread per (b,d), 16-state in regs ----------
__global__ void scan_spatial_kernel(const float* __restrict__ xs,    // (BL,DI)
                                    const float* __restrict__ delta, // (BL,DI)
                                    const float* __restrict__ xdbl,  // (BL,48): [dt|B|C]
                                    const float* __restrict__ A_logs,// (DI,16)
                                    const float* __restrict__ Ds,    // (DI)
                                    float* __restrict__ y)           // (BL,DI)
{
    int i = blockIdx.x * blockDim.x + threadIdx.x;
    if (i >= cBD) return;
    int d = i % cDI, b = i / cDI;
    float a[cN], h[cN];
    #pragma unroll
    for (int n = 0; n < cN; ++n) { a[n] = -__expf(A_logs[d * cN + n]); h[n] = 0.f; }
    float Dp = Ds[d];
    for (int t = 0; t < cL; ++t) {
        size_t r = (size_t)(b * cL + t);
        float dlt = delta[r * cDI + d];
        float xv  = xs[r * cDI + d];
        float dx  = dlt * xv;
        const float* bc = xdbl + r * 48;
        float yv = 0.f;
        #pragma unroll
        for (int n = 0; n < cN; ++n) {
            h[n] = __expf(dlt * a[n]) * h[n] + dx * bc[16 + n];
            yv  += h[n] * bc[32 + n];
        }
        y[r * cDI + d] = yv + Dp * xv;
    }
}

// ---------- pooling helpers ----------
__global__ void row_mean_kernel(const float* __restrict__ X, float* __restrict__ out,
                                int rows, int cols)
{
    int r = blockIdx.x * blockDim.x + threadIdx.x;
    if (r >= rows) return;
    float s = 0.f;
    for (int c = 0; c < cols; ++c) s += X[(size_t)r * cols + c];
    out[r] = s / cols;
}

__global__ void col_mean_kernel(const float* __restrict__ X, float* __restrict__ out)
{   // out[b*DI+d] = mean_l X[(b*L+l)*DI + d]
    int i = blockIdx.x * blockDim.x + threadIdx.x;
    if (i >= cBD) return;
    int d = i % cDI, b = i / cDI;
    float s = 0.f;
    for (int l = 0; l < cL; ++l) s += X[((size_t)(b * cL + l)) * cDI + d];
    out[i] = s / cL;
}

// ---------- parameter-free cross-sample attention (32x32) ----------
__global__ void interatt_scores_kernel(const float* __restrict__ pool, int Cv,
                                       float* __restrict__ sval, int* __restrict__ sidx)
{
    int b = threadIdx.x;                 // one wave, 32 threads
    __shared__ float norms[32];
    float nn = 0.f;
    for (int t = 0; t < Cv; ++t) { float v = pool[b * Cv + t]; nn += v * v; }
    norms[b] = fmaxf(sqrtf(nn), 1e-12f);
    __syncthreads();
    float best = 0.f; int bi = b;        // diagonal is zeroed in reference
    for (int j = 0; j < 32; ++j) {
        if (j == b) continue;
        float dq = 0.f;
        for (int t = 0; t < Cv; ++t) dq += pool[b * Cv + t] * pool[j * Cv + t];
        dq /= (norms[b] * norms[j]);
        if (dq > best) { best = dq; bi = j; }
    }
    sval[b] = best; sidx[b] = bi;
}

// ---------- LayerNorm(inter-att shifted) * silu(z) ----------
__global__ void ln_gate_kernel(const float* __restrict__ y, const float* __restrict__ pool,
                               const float* __restrict__ sval, const int* __restrict__ sidx,
                               const float* __restrict__ gamma, const float* __restrict__ beta,
                               const float* __restrict__ xz, float* __restrict__ g)
{
    int row = blockIdx.x;                // (b,t)
    int b = row / cL, t = row % cL;
    float cadd = sval[b] * pool[sidx[b] * cL + t];   // uniform over d
    __shared__ float red[256];
    __shared__ float mu_s, rstd_s;
    float s = 0.f;
    for (int d = threadIdx.x; d < cDI; d += 256) s += y[(size_t)row * cDI + d] + cadd;
    red[threadIdx.x] = s; __syncthreads();
    for (int st = 128; st > 0; st >>= 1) {
        if (threadIdx.x < st) red[threadIdx.x] += red[threadIdx.x + st];
        __syncthreads();
    }
    if (threadIdx.x == 0) mu_s = red[0] / cDI;
    __syncthreads();
    float mu = mu_s, v = 0.f;
    for (int d = threadIdx.x; d < cDI; d += 256) {
        float e = y[(size_t)row * cDI + d] + cadd - mu; v += e * e;
    }
    red[threadIdx.x] = v; __syncthreads();
    for (int st = 128; st > 0; st >>= 1) {
        if (threadIdx.x < st) red[threadIdx.x] += red[threadIdx.x + st];
        __syncthreads();
    }
    if (threadIdx.x == 0) rstd_s = rsqrtf(red[0] / cDI + 1e-5f);
    __syncthreads();
    float rstd = rstd_s;
    for (int d = threadIdx.x; d < cDI; d += 256) {
        float val = ((y[(size_t)row * cDI + d] + cadd) - mu) * rstd * gamma[d] + beta[d];
        float z = xz[(size_t)row * (2 * cDI) + cDI + d];
        g[(size_t)row * cDI + d] = val * siluf(z);
    }
}

// ---------- spectral adaptive attention ----------
__global__ void spec_meanmax_kernel(const float* __restrict__ xc,
                                    float* __restrict__ meanv, float* __restrict__ maxv)
{
    int i = blockIdx.x * blockDim.x + threadIdx.x;
    if (i >= cBD) return;
    int d = i % cDI, b = i / cDI;
    float s = 0.f, mx = -1e30f;
    for (int l = 0; l < cL; ++l) {
        float v = xc[((size_t)(b * cL + l)) * cDI + d];
        s += v; mx = fmaxf(mx, v);
    }
    meanv[i] = s / cL; maxv[i] = mx;
}

__global__ void spec_mlp_kernel(const float* __restrict__ meanv, const float* __restrict__ maxv,
                                const float* __restrict__ fc1, float* __restrict__ hs)
{
    int b = blockIdx.x;
    __shared__ float r1[256], r2[256];
    float s1 = 0.f, s2 = 0.f;
    for (int d = threadIdx.x; d < cDI; d += 256) {
        float f = fc1[d];
        s1 += meanv[b * cDI + d] * f;
        s2 += maxv[b * cDI + d] * f;
    }
    r1[threadIdx.x] = s1; r2[threadIdx.x] = s2;
    __syncthreads();
    for (int st = 128; st > 0; st >>= 1) {
        if (threadIdx.x < st) { r1[threadIdx.x] += r1[threadIdx.x + st]; r2[threadIdx.x] += r2[threadIdx.x + st]; }
        __syncthreads();
    }
    if (threadIdx.x == 0) hs[b] = fmaxf(r1[0], 0.f) + fmaxf(r2[0], 0.f);
}

__global__ void spec_apply_kernel(const float* __restrict__ hs, const float* __restrict__ fc2,
                                  const float* __restrict__ am, float* __restrict__ xc)
{
    int i = blockIdx.x * blockDim.x + threadIdx.x;
    if (i >= cBL * cDI) return;
    int d = i % cDI, b = i / (cL * cDI);
    float att = sigf(hs[b] * fc2[d]);
    xc[i] *= sigf(att * (1.0f + am[d]));
}

__global__ void transpose_spec_kernel(const float* __restrict__ in, float* __restrict__ out)
{   // out[(b*DI+t)*L + l] = in[(b*L+l)*DI + t]
    int i = blockIdx.x * blockDim.x + threadIdx.x;
    if (i >= cB * cDI * cL) return;
    int l = i % cL;
    int t = (i / cL) % cDI;
    int b = i / (cL * cDI);
    out[i] = in[((size_t)(b * cL + l)) * cDI + t];
}

// ---------- spectral bidirectional scan: one thread per (b,l) ----------
__global__ void scan_spec_kernel(const float* __restrict__ xscan,  // (B,L,DI)
                                 const float* __restrict__ delta0, // (B*DI, L)
                                 const float* __restrict__ delta1,
                                 const float* __restrict__ dbl0,   // (B*DI, 48)
                                 const float* __restrict__ dbl1,
                                 const float* __restrict__ A_logs, // (2L,16)
                                 const float* __restrict__ Dsp,    // (2L)
                                 float* __restrict__ ys)           // (B,L,DI)
{
    int i = blockIdx.x * blockDim.x + threadIdx.x;
    if (i >= cBL) return;
    int l = i % cL, b = i / cL;
    const float* xrow = xscan + (size_t)i * cDI;
    float* yrow = ys + (size_t)i * cDI;
    float Dsum = Dsp[l] + Dsp[cL + l];
    float a[cN], h[cN];

    // k = 0 : forward over t
    #pragma unroll
    for (int n = 0; n < cN; ++n) { a[n] = -__expf(A_logs[l * cN + n]); h[n] = 0.f; }
    for (int t = 0; t < cDI; ++t) {
        size_t r = (size_t)(b * cDI + t);
        float dlt = delta0[r * cL + l];
        float xv  = xrow[t];
        float dx  = dlt * xv;
        const float* bc = dbl0 + r * 48;
        float yv = 0.f;
        #pragma unroll
        for (int n = 0; n < cN; ++n) {
            h[n] = __expf(dlt * a[n]) * h[n] + dx * bc[16 + n];
            yv  += h[n] * bc[32 + n];
        }
        yrow[t] = yv + Dsum * xv;
    }
    // k = 1 : reversed over t (flip in, flip out)
    #pragma unroll
    for (int n = 0; n < cN; ++n) { a[n] = -__expf(A_logs[(cL + l) * cN + n]); h[n] = 0.f; }
    for (int tt = 0; tt < cDI; ++tt) {
        int t = cDI - 1 - tt;
        size_t r = (size_t)(b * cDI + t);
        float dlt = delta1[r * cL + l];
        float xv  = xrow[t];
        float dx  = dlt * xv;
        const float* bc = dbl1 + r * 48;
        float yv = 0.f;
        #pragma unroll
        for (int n = 0; n < cN; ++n) {
            h[n] = __expf(dlt * a[n]) * h[n] + dx * bc[16 + n];
            yv  += h[n] * bc[32 + n];
        }
        yrow[t] += yv;
    }
}

// ---------- BN (eval, running stats 0/1) + inter-att + silu(z) gate ----------
__global__ void bn_gate_kernel(const float* __restrict__ ys, const float* __restrict__ pool,
                               const float* __restrict__ sval, const int* __restrict__ sidx,
                               const float* __restrict__ gamma, const float* __restrict__ beta,
                               const float* __restrict__ xz, float* __restrict__ g)
{
    int i = blockIdx.x * blockDim.x + threadIdx.x;
    if (i >= cBL * cDI) return;
    int d = i % cDI;
    int row = i / cDI;                    // b*L + l
    int b = row / cL;
    const float inv = 0.99999500003749937f;   // 1/sqrt(1+1e-5)
    float v = ys[i] + sval[b] * pool[sidx[b] * cDI + d];
    v = v * inv * gamma[d] + beta[d];
    float z = xz[(size_t)row * (2 * cDI) + cDI + d];
    g[i] = v * siluf(z);
}

__global__ void combine_kernel(const float* __restrict__ oq, const float* __restrict__ os,
                               const float* __restrict__ sw, float* __restrict__ out, int n)
{
    int i = blockIdx.x * blockDim.x + threadIdx.x;
    if (i >= n) return;
    float w0 = sw[0], w1 = sw[1];
    out[i] = (w0 * oq[i] + w1 * os[i]) / (w0 + w1);
}

// =====================================================================
// host side
// =====================================================================
extern "C" void kernel_launch(void* const* d_in, const int* in_sizes, int n_in,
                              void* d_out, int out_size, void* d_ws, size_t ws_size,
                              hipStream_t stream)
{
    (void)in_sizes; (void)n_in; (void)out_size; (void)ws_size;

    const float* x            = (const float*)d_in[0];
    const float* in_q_w       = (const float*)d_in[1];
    const float* in_s_w       = (const float*)d_in[2];
    const float* conv_q_w     = (const float*)d_in[3];
    const float* conv_q_b     = (const float*)d_in[4];
    const float* conv_s_w     = (const float*)d_in[5];
    const float* conv_s_b     = (const float*)d_in[6];
    const float* spa_conv_w   = (const float*)d_in[7];
    const float* spa_aw       = (const float*)d_in[8];
    const float* spec_fc1     = (const float*)d_in[9];
    const float* spec_fc2     = (const float*)d_in[10];
    const float* spec_am      = (const float*)d_in[11];
    const float* x_proj_w     = (const float*)d_in[12];
    const float* dt_proj_w    = (const float*)d_in[13];
    const float* dt_proj_b    = (const float*)d_in[14];
    const float* A_logs       = (const float*)d_in[15];
    const float* Ds           = (const float*)d_in[16];
    const float* x_proj_ws    = (const float*)d_in[17];
    const float* dt_proj_ws   = (const float*)d_in[18];
    const float* dt_proj_bs   = (const float*)d_in[19];
    const float* A_logs_sp    = (const float*)d_in[20];
    const float* Ds_sp        = (const float*)d_in[21];
    const float* ln_gamma     = (const float*)d_in[22];
    const float* ln_beta      = (const float*)d_in[23];
    const float* bn_gamma     = (const float*)d_in[24];
    const float* bn_beta      = (const float*)d_in[25];
    const float* out_q_w      = (const float*)d_in[26];
    const float* out_s_w      = (const float*)d_in[27];
    const float* sum_w        = (const float*)d_in[28];
    float* out = (float*)d_out;

    // ---- workspace layout (floats); phase-1 / phase-2 regions are reused ----
    float* ws = (float*)d_ws;
    const size_t XZ    = 0;                               // 7200*1024
    const size_t CONV  = XZ    + (size_t)cBL * 2 * cDI;   // 7200*512
    const size_t DBL   = CONV  + (size_t)cBL * cDI;       // 2 * 16384*48
    const size_t DELTA = DBL   + (size_t)cBD * 48 * 2;    // 2 * 16384*225
    const size_t YB    = DELTA + (size_t)cBD * cL * 2;    // 7200*512 (y_q / xscanT / g_s)
    const size_t GB    = YB    + (size_t)cBL * cDI;       // 7200*512 (g_q / ys)
    const size_t OUTQ  = GB    + (size_t)cBL * cDI;       // 7200*256
    const size_t OUTS  = OUTQ  + (size_t)cBL * cC;        // 7200*256
    const size_t SMALL = OUTS  + (size_t)cBL * cC;
    float* mm    = ws + SMALL;                 // 2*16384 (mm / meanv+maxv)
    float* pool  = ws + SMALL + 32768;         // 16384
    float* sval  = ws + SMALL + 49152;         // 32
    int*   sidx  = (int*)(ws + SMALL + 49184); // 32
    float* hs    = ws + SMALL + 49216;         // 32

    auto ew = [](int n) { return dim3((n + 255) / 256); };
    auto gg = [](int M, int N) { return dim3((N + 15) / 16, (M + 63) / 64); };

    // ================= phase 1: spatial branch =================
    wmma_gemm_kernel<256, true, true><<<gg(cBL, 2 * cDI), 128, 0, stream>>>(
        x, cC, in_q_w, cC, ws + XZ, 2 * cDI, cBL, 2 * cDI);                      // xz_q
    dwconv_silu_kernel<<<ew(cBL * cDI), 256, 0, stream>>>(ws + XZ, conv_q_w, conv_q_b, ws + CONV);
    spa_meanmax_kernel<<<cBL, 256, 0, stream>>>(ws + CONV, mm);
    spa_apply_kernel<<<cBL, 256, 0, stream>>>(mm, spa_conv_w, spa_aw, ws + CONV);
    wmma_gemm_kernel<512, true, true><<<gg(cBL, 48), 128, 0, stream>>>(
        ws + CONV, cDI, x_proj_w, cDI, ws + DBL, 48, cBL, 48);                   // x_dbl
    wmma_gemm_kernel<16, false, false><<<gg(cBL, cDI), 128, 0, stream>>>(
        ws + DBL, 48, dt_proj_w, 16, ws + DELTA, cDI, cBL, cDI);                 // delta raw
    softplus_bias_kernel<<<ew(cBL * cDI), 256, 0, stream>>>(ws + DELTA, dt_proj_b, cBL, cDI);
    scan_spatial_kernel<<<ew(cBD), 256, 0, stream>>>(ws + CONV, ws + DELTA, ws + DBL,
                                                     A_logs, Ds, ws + YB);
    row_mean_kernel<<<ew(cBL), 256, 0, stream>>>(ws + YB, pool, cBL, cDI);
    interatt_scores_kernel<<<1, 32, 0, stream>>>(pool, cL, sval, sidx);
    ln_gate_kernel<<<cBL, 256, 0, stream>>>(ws + YB, pool, sval, sidx,
                                            ln_gamma, ln_beta, ws + XZ, ws + GB);
    wmma_gemm_kernel<512, true, true><<<gg(cBL, cC), 128, 0, stream>>>(
        ws + GB, cDI, out_q_w, cDI, ws + OUTQ, cC, cBL, cC);                     // out_q

    // ================= phase 2: spectral branch =================
    wmma_gemm_kernel<256, true, true><<<gg(cBL, 2 * cDI), 128, 0, stream>>>(
        x, cC, in_s_w, cC, ws + XZ, 2 * cDI, cBL, 2 * cDI);                      // xz_s
    dwconv_silu_kernel<<<ew(cBL * cDI), 256, 0, stream>>>(ws + XZ, conv_s_w, conv_s_b, ws + CONV);
    spec_meanmax_kernel<<<ew(cBD), 256, 0, stream>>>(ws + CONV, mm, mm + cBD);
    spec_mlp_kernel<<<cB, 256, 0, stream>>>(mm, mm + cBD, spec_fc1, hs);
    spec_apply_kernel<<<ew(cBL * cDI), 256, 0, stream>>>(hs, spec_fc2, spec_am, ws + CONV);
    transpose_spec_kernel<<<ew(cB * cDI * cL), 256, 0, stream>>>(ws + CONV, ws + YB);

    float* dbl0 = ws + DBL;
    float* dbl1 = ws + DBL + (size_t)cBD * 48;
    float* dl0  = ws + DELTA;
    float* dl1  = ws + DELTA + (size_t)cBD * cL;
    wmma_gemm_kernel<225, false, false><<<gg(cBD, 48), 128, 0, stream>>>(
        ws + YB, cL, x_proj_ws, cL, dbl0, 48, cBD, 48);
    wmma_gemm_kernel<225, false, false><<<gg(cBD, 48), 128, 0, stream>>>(
        ws + YB, cL, x_proj_ws + 48 * cL, cL, dbl1, 48, cBD, 48);
    wmma_gemm_kernel<16, false, false><<<gg(cBD, cL), 128, 0, stream>>>(
        dbl0, 48, dt_proj_ws, 16, dl0, cL, cBD, cL);
    wmma_gemm_kernel<16, false, false><<<gg(cBD, cL), 128, 0, stream>>>(
        dbl1, 48, dt_proj_ws + cL * 16, 16, dl1, cL, cBD, cL);
    softplus_bias_kernel<<<ew(cBD * cL), 256, 0, stream>>>(dl0, dt_proj_bs,      cBD, cL);
    softplus_bias_kernel<<<ew(cBD * cL), 256, 0, stream>>>(dl1, dt_proj_bs + cL, cBD, cL);

    scan_spec_kernel<<<ew(cBL), 256, 0, stream>>>(ws + CONV, dl0, dl1, dbl0, dbl1,
                                                  A_logs_sp, Ds_sp, ws + GB);
    col_mean_kernel<<<ew(cBD), 256, 0, stream>>>(ws + GB, pool);
    interatt_scores_kernel<<<1, 32, 0, stream>>>(pool, cDI, sval, sidx);
    bn_gate_kernel<<<ew(cBL * cDI), 256, 0, stream>>>(ws + GB, pool, sval, sidx,
                                                      bn_gamma, bn_beta, ws + XZ, ws + YB);
    wmma_gemm_kernel<512, true, true><<<gg(cBL, cC), 128, 0, stream>>>(
        ws + YB, cDI, out_s_w, cDI, ws + OUTS, cC, cBL, cC);                     // out_s

    combine_kernel<<<ew(cBL * cC), 256, 0, stream>>>(ws + OUTQ, ws + OUTS, sum_w,
                                                     out, cBL * cC);
}
